// PnPNystraAttention_23673859735833
// MI455X (gfx1250) — matrix operand order repaired and checked by reading, not compiled, not run on hardware
//
#include <hip/hip_runtime.h>
#include <hip/hip_bf16.h>

typedef __bf16 bf16;
typedef __attribute__((ext_vector_type(16))) __bf16 v16bf;
typedef __attribute__((ext_vector_type(8)))  __bf16 v8bf;
typedef __attribute__((ext_vector_type(8)))  float  v8f;
typedef __attribute__((ext_vector_type(2)))  float  v2f;
typedef __attribute__((ext_vector_type(4)))  int    v4i;

#define BB 64
#define NN 1024
#define CC 512
#define HH 8
#define DD 64
#define MM 64
#define BN (BB*NN)
#define BHN (BB*HH)
#define D1P 80
#define QSCALE 0.125f   // 64^-0.5

// async-LDS builtins exist on this toolchain (probe-verified via __has_builtin)
#if defined(__has_builtin)
#if __has_builtin(__builtin_amdgcn_global_load_async_to_lds_b128) && \
    __has_builtin(__builtin_amdgcn_s_wait_asynccnt)
#define USE_ASYNC_LDS 1
#endif
#endif

#define ASG __attribute__((address_space(1)))
#define ASL __attribute__((address_space(3)))

// 16-byte global -> LDS copy through the ASYNCcnt path when available.
__device__ __forceinline__ void async_cp16(const void* g, void* l) {
#ifdef USE_ASYNC_LDS
    __builtin_amdgcn_global_load_async_to_lds_b128((ASG v4i*)g, (ASL v4i*)l, 0, 0);
#else
    *reinterpret_cast<uint4*>(l) = *reinterpret_cast<const uint4*>(g);
#endif
}
__device__ __forceinline__ void async_wait() {
#ifdef USE_ASYNC_LDS
    __builtin_amdgcn_s_wait_asynccnt(0);
#endif
}

__device__ __forceinline__ float activf(float x) {
    return __expf(fminf(x, 0.0f)) + fmaxf(x, 0.0f);
}

// ---- WMMA fragment loaders (ISA 7.12.2, wave32), fully vectorized ---------

// A 16x32 bf16, row-major in LDS: elements 0..15 = K(half*16 .. +15)
// contiguous: two aligned 16B LDS loads.
__device__ __forceinline__ v16bf ldsA16(const bf16* A, int lda, int row0, int k0, int lane) {
    int half = lane >> 4, m = lane & 15;
    const bf16* r = A + (row0 + m) * lda + k0 + half * 16;
    v8bf lo = *reinterpret_cast<const v8bf*>(r);
    v8bf hi = *reinterpret_cast<const v8bf*>(r + 8);
    return __builtin_shufflevector(lo, hi, 0,1,2,3,4,5,6,7,8,9,10,11,12,13,14,15);
}

// B 32x16 bf16 from K-contiguous ("Bt" = B transposed, [n][k]) LDS storage:
// elements 0..15 = K(half*16 .. +15) contiguous: two aligned 16B LDS loads.
__device__ __forceinline__ v16bf ldsB16(const bf16* Bt, int ldbt, int col0, int k0, int lane) {
    int half = lane >> 4, n = lane & 15;
    const bf16* r = Bt + (col0 + n) * ldbt + k0 + half * 16;
    v8bf lo = *reinterpret_cast<const v8bf*>(r);
    v8bf hi = *reinterpret_cast<const v8bf*>(r + 8);
    return __builtin_shufflevector(lo, hi, 0,1,2,3,4,5,6,7,8,9,10,11,12,13,14,15);
}

// ---- elementwise conversion / init kernels --------------------------------

__global__ void f2bf_kernel(const float* __restrict__ in, bf16* __restrict__ out, long n) {
    long i = (long)blockIdx.x * blockDim.x + threadIdx.x;
    if (i < n) out[i] = (bf16)in[i];
}

// bf16 cast + transpose: out[c*rows + r] = in[r*cols + c]
__global__ void f2bf_t_kernel(const float* __restrict__ in, bf16* __restrict__ out,
                              int rows, int cols) {
    long i = (long)blockIdx.x * blockDim.x + threadIdx.x;
    if (i >= (long)rows * cols) return;
    int r = (int)(i / cols), c = (int)(i % cols);
    out[(size_t)c * rows + r] = (bf16)in[i];
}

// v1 stored transposed: [BH][80][NN]; rows 64..79 are the ones/zero padding.
__global__ void init_v1pad_kernel(bf16* __restrict__ v1t) {
    long i = (long)blockIdx.x * blockDim.x + threadIdx.x;
    if (i >= (long)BHN * 16 * NN) return;
    long n = i % NN;
    long rest = i / NN;
    int c16 = (int)(rest % 16);
    long bh = rest / 16;
    v1t[((size_t)bh * D1P + 64 + c16) * NN + n] = (c16 == 0) ? (bf16)1.0f : (bf16)0.0f;
}

__global__ void zero_red_kernel(unsigned* red) { red[0] = 0u; red[1] = 0u; }

// ---- projection GEMM: Y = X @ W + b (W pre-transposed bf16) ---------------
// MODE 0: Q  -> qb [B,H,N,D] bf16 * QSCALE
// MODE 1: KV -> kb [B,H,N,D] bf16 ; v1t [B,H,80,N] bf16
// MODE 2: out -> out_f [BN,512] fp32
template <int MODE>
__global__ __launch_bounds__(256) void gemm_proj(
    const bf16* __restrict__ X, const bf16* __restrict__ Wt,
    const float* __restrict__ bias, int K,
    bf16* __restrict__ out_q, bf16* __restrict__ out_k,
    bf16* __restrict__ out_v1, float* __restrict__ out_f)
{
    __shared__ bf16 As[64][40];   // [row][k]   row stride 80B (16B aligned)
    __shared__ bf16 Bt[64][40];   // [col][k]   K-contiguous for B fragments
    const int tid  = threadIdx.x;
    const int lane = tid & 31, wave = tid >> 5;
    const int half = lane >> 4, ln = lane & 15;
    const int row0 = blockIdx.y * 64;
    const int col0 = blockIdx.x * 64;
    const int r4 = tid >> 2, q4 = tid & 3;

    v8f acc[2] = {};
    for (int k0 = 0; k0 < K; k0 += 32) {
        const bf16* gx = &X [(size_t)(row0 + r4) * K + k0 + q4 * 8];
        const bf16* gw = &Wt[(size_t)(col0 + r4) * K + k0 + q4 * 8];
        if (k0 + 32 < K) {
            __builtin_prefetch(gx + 32, 0, 1);
            __builtin_prefetch(gw + 32, 0, 1);
        }
        async_cp16(gx, &As[r4][q4 * 8]);
        async_cp16(gw, &Bt[r4][q4 * 8]);
        async_wait();
        __syncthreads();
#pragma unroll
        for (int ti = 0; ti < 2; ++ti) {
            int t = wave * 2 + ti, tr = t >> 2, tc = t & 3;
            v16bf a = ldsA16(&As[0][0], 40, tr * 16, 0, lane);
            v16bf b = ldsB16(&Bt[0][0], 40, tc * 16, 0, lane);
            acc[ti] = __builtin_amdgcn_wmma_f32_16x16x32_bf16(
                false, a, false, b, (short)0, acc[ti], false, false);
        }
        __syncthreads();
    }

#pragma unroll
    for (int ti = 0; ti < 2; ++ti) {
        int t = wave * 2 + ti, tr = t >> 2, tc = t & 3;
        int col = col0 + tc * 16 + ln;
        float bv = bias[col];
#pragma unroll
        for (int r = 0; r < 8; ++r) {
            int row = row0 + tr * 16 + half * 8 + r;
            float val = acc[ti][r] + bv;
            int b = row >> 10, n = row & 1023;
            if (MODE == 0) {
                int h = col >> 6, d = col & 63;
                out_q[(((size_t)(b * HH + h)) * NN + n) * DD + d] = (bf16)(val * QSCALE);
            } else if (MODE == 1) {
                if (col < CC) {
                    int h = col >> 6, d = col & 63;
                    out_k[(((size_t)(b * HH + h)) * NN + n) * DD + d] = (bf16)val;
                } else {
                    int c2 = col - CC, h = c2 >> 6, d = c2 & 63;
                    out_v1[(((size_t)(b * HH + h)) * D1P + d) * NN + n] = (bf16)val;
                }
            } else {
                out_f[(size_t)row * CC + col] = val;
            }
        }
    }
}

// ---- landmark pooling: exact 4x4 block means on the 32x32 grid ------------
__global__ void pool_lm_kernel(const bf16* __restrict__ in, bf16* __restrict__ out) {
    long idx = (long)blockIdx.x * blockDim.x + threadIdx.x;  // BH*M*D
    if (idx >= (long)BHN * MM * DD) return;
    int d = idx & 63;
    int m = (idx >> 6) & 63;
    long bh = idx >> 12;
    int i = m >> 3, j = m & 7;
    float s = 0.0f;
#pragma unroll
    for (int a = 0; a < 4; ++a)
#pragma unroll
        for (int c = 0; c < 4; ++c) {
            int n = i * 128 + a * 32 + j * 4 + c;
            s += (float)in[(bh * NN + n) * DD + d];
        }
    out[idx] = (bf16)(s * 0.0625f);
}

// ---- temp = activ(q_m @ k_m^T); global abs row/col-sum max ----------------
__global__ __launch_bounds__(256) void temp_kernel(
    const bf16* __restrict__ qmb, const bf16* __restrict__ kmb,
    float* __restrict__ temp, unsigned* __restrict__ red)
{
    int bh = blockIdx.x;
    __shared__ bf16 qs[64][72];
    __shared__ bf16 kmB[64][72];
    __shared__ float tS[64][68];
    const bf16* qg = qmb + (size_t)bh * 4096;
    const bf16* kg = kmb + (size_t)bh * 4096;
    int tid = threadIdx.x;
    for (int e = tid; e < 512; e += 256) {
        int r = e >> 3, q = e & 7;
        async_cp16(qg + r * 64 + q * 8, &qs[r][q * 8]);
        async_cp16(kg + r * 64 + q * 8, &kmB[r][q * 8]);
    }
    async_wait();
    __syncthreads();
    int lane = tid & 31, wave = tid >> 5, half = lane >> 4, ln = lane & 15;
    float* tg = temp + (size_t)bh * 4096;
#pragma unroll
    for (int ti = 0; ti < 2; ++ti) {
        int t = wave * 2 + ti, tr = t >> 2, tc = t & 3;
        v8f c = {};
#pragma unroll
        for (int k0 = 0; k0 < 64; k0 += 32) {
            v16bf a = ldsA16(&qs[0][0], 72, tr * 16, k0, lane);
            v16bf b = ldsB16(&kmB[0][0], 72, tc * 16, k0, lane);
            c = __builtin_amdgcn_wmma_f32_16x16x32_bf16(false, a, false, b, (short)0, c, false, false);
        }
#pragma unroll
        for (int r = 0; r < 8; ++r) {
            int row = tr * 16 + half * 8 + r, col = tc * 16 + ln;
            float v = activf(c[r]);
            tS[row][col] = v;
            tg[row * 64 + col] = v;
        }
    }
    __syncthreads();
    if (tid < 64) {
        float rs = 0.0f, cs = 0.0f;
        for (int j = 0; j < 64; ++j) { rs += fabsf(tS[tid][j]); cs += fabsf(tS[j][tid]); }
        atomicMax(&red[0], __float_as_uint(rs));
        atomicMax(&red[1], __float_as_uint(cs));
    }
}

// ---- 64x64 f32 matmul: D = scale*(A @ B) with Bt (transposed B) in LDS ----
__device__ __forceinline__ void mm64_f32(float* D, float* Dt,
                                         const float* A, const float* Bt,
                                         int ld, float scale, int tid)
{
    int lane = tid & 31, wave = tid >> 5;
    int half = lane >> 4, ln = lane & 15;
#pragma unroll
    for (int ti = 0; ti < 2; ++ti) {
        int t = wave * 2 + ti, tr = t >> 2, tc = t & 3;
        v8f c = {};
#pragma unroll
        for (int k0 = 0; k0 < 64; k0 += 4) {
            v2f a = *reinterpret_cast<const v2f*>(A  + (tr * 16 + ln) * ld + k0 + half * 2);
            v2f b = *reinterpret_cast<const v2f*>(Bt + (tc * 16 + ln) * ld + k0 + half * 2);
            c = __builtin_amdgcn_wmma_f32_16x16x4_f32(false, a, false, b, (short)0, c, false, false);
        }
#pragma unroll
        for (int r = 0; r < 8; ++r) {
            float v = c[r] * scale;
            int row = tr * 16 + half * 8 + r, col = tc * 16 + ln;
            D[row * ld + col] = v;
            if (Dt) Dt[col * ld + row] = v;
        }
    }
}

// ---- Moore-Penrose iterative pinv, one block per head ---------------------
__global__ __launch_bounds__(256) void pinv_kernel(
    const float* __restrict__ temp, const float* __restrict__ redf,
    float* __restrict__ pinvz)
{
    extern __shared__ float smem[];
    const int LD = 68, SZ = 64 * 68;
    float* Xs = smem;            // X, A-role only (row-major)
    float* Z  = smem + SZ;       // Z row-major (A of final mm)
    float* U  = smem + 2 * SZ;   // scratch / next Z row-major
    float* Zt = smem + 3 * SZ;   // Z transposed (B of X@Z)
    float* Tt = smem + 4 * SZ;   // (cI - xz) transposed (B-role)
    float* XZ = smem + 5 * SZ;   // xz row-major (A-role)

    int bh = blockIdx.x, tid = threadIdx.x;
    const float* g = temp + (size_t)bh * 4096;
    float inv = 1.0f / (redf[0] * redf[1]);
    for (int e = tid; e < 1024; e += 256) {
        int r = e >> 4, q = e & 15;
        float4 v4 = *reinterpret_cast<const float4*>(g + r * 64 + q * 4);
        *reinterpret_cast<float4*>(Xs + r * LD + q * 4) = v4;
        int c = q * 4;
        Z[(c + 0) * LD + r] = v4.x * inv;  Zt[r * LD + c + 0] = v4.x * inv;
        Z[(c + 1) * LD + r] = v4.y * inv;  Zt[r * LD + c + 1] = v4.y * inv;
        Z[(c + 2) * LD + r] = v4.z * inv;  Zt[r * LD + c + 2] = v4.z * inv;
        Z[(c + 3) * LD + r] = v4.w * inv;  Zt[r * LD + c + 3] = v4.w * inv;
    }
    __syncthreads();

    for (int it = 0; it < 6; ++it) {
        mm64_f32(XZ, nullptr, Xs, Zt, LD, 1.0f, tid); __syncthreads();
        for (int e = tid; e < 4096; e += 256) { int r = e >> 6, c = e & 63;
            float v = -XZ[r * LD + c]; if (r == c) v += 7.0f; Tt[c * LD + r] = v; }
        __syncthreads();
        mm64_f32(U, nullptr, XZ, Tt, LD, 1.0f, tid); __syncthreads();
        for (int e = tid; e < 4096; e += 256) { int r = e >> 6, c = e & 63;
            float v = -U[r * LD + c]; if (r == c) v += 15.0f; Tt[c * LD + r] = v; }
        __syncthreads();
        mm64_f32(U, nullptr, XZ, Tt, LD, 1.0f, tid); __syncthreads();
        for (int e = tid; e < 4096; e += 256) { int r = e >> 6, c = e & 63;
            float v = -U[r * LD + c]; if (r == c) v += 13.0f; Tt[c * LD + r] = v; }
        __syncthreads();
        // newZ = 0.25 * Z @ T ; row-major into U, transposed into Zt
        mm64_f32(U, Zt, Z, Tt, LD, 0.25f, tid); __syncthreads();
        float* tp = Z; Z = U; U = tp;
    }
    float* o = pinvz + (size_t)bh * 4096;
    for (int e = tid; e < 4096; e += 256) { int r = e >> 6, c = e & 63; o[e] = Z[r * LD + c]; }
}

// ---- kern_qmk_v = activ(q_m @ k^T) @ v1 -> kqv_t [BH][80][64] fp32 --------
__global__ __launch_bounds__(256) void kqmv_kernel(
    const bf16* __restrict__ qmb, const bf16* __restrict__ kb,
    const bf16* __restrict__ v1t, float* __restrict__ kqv_t)
{
    int bh = blockIdx.x;
    __shared__ bf16 qm[64][72];
    __shared__ bf16 ktB[64][72];   // k chunk row-major == Bt of k^T
    __shared__ bf16 Sb[64][72];    // activ result, A-role of second GEMM
    __shared__ bf16 Vt[80][72];    // v1^T chunk: [c][n], Bt of v1
    int tid = threadIdx.x, lane = tid & 31, wave = tid >> 5;
    int half = lane >> 4, ln = lane & 15;

    const bf16* qg = qmb + (size_t)bh * 4096;
    for (int e = tid; e < 512; e += 256) {
        int r = e >> 3, q = e & 7;
        async_cp16(qg + r * 64 + q * 8, &qm[r][q * 8]);
    }

    v8f acc[3] = {};
    for (int ch = 0; ch < 16; ++ch) {
        __syncthreads();
        const bf16* kg = kb + ((size_t)bh * NN + ch * 64) * DD;
        for (int e = tid; e < 512; e += 256) {
            int r = e >> 3, q = e & 7;
            async_cp16(kg + r * 64 + q * 8, &ktB[r][q * 8]);
        }
        for (int e = tid; e < 640; e += 256) {
            int r = e >> 3, q = e & 7;
            async_cp16(v1t + ((size_t)bh * D1P + r) * NN + ch * 64 + q * 8, &Vt[r][q * 8]);
        }
        async_wait();
        __syncthreads();

        // S = activ(qm @ k_chunk^T), staged bf16 for second GEMM
#pragma unroll
        for (int ti = 0; ti < 2; ++ti) {
            int t = wave * 2 + ti, tr = t >> 2, tc = t & 3;
            v8f c = {};
#pragma unroll
            for (int k0 = 0; k0 < 64; k0 += 32) {
                v16bf a = ldsA16(&qm[0][0], 72, tr * 16, k0, lane);
                v16bf b = ldsB16(&ktB[0][0], 72, tc * 16, k0, lane);
                c = __builtin_amdgcn_wmma_f32_16x16x32_bf16(false, a, false, b, (short)0, c, false, false);
            }
#pragma unroll
            for (int r = 0; r < 8; ++r)
                Sb[tr * 16 + half * 8 + r][tc * 16 + ln] = (bf16)activf(c[r]);
        }
        __syncthreads();

        // acc += S @ v1_chunk (tiles: 4 rows x 5 cols over [64,80])
        int ai = 0;
        for (int t = wave; t < 20; t += 8, ++ai) {
            int tr = t / 5, tc = t % 5;
#pragma unroll
            for (int k0 = 0; k0 < 64; k0 += 32) {
                v16bf a = ldsA16(&Sb[0][0], 72, tr * 16, k0, lane);
                v16bf b = ldsB16(&Vt[0][0], 72, tc * 16, k0, lane);
                acc[ai] = __builtin_amdgcn_wmma_f32_16x16x32_bf16(false, a, false, b, (short)0, acc[ai], false, false);
            }
        }
    }
    // store transposed: kqv_t[c][m]
    float* og = kqv_t + (size_t)bh * D1P * 64;
    int ai = 0;
    for (int t = wave; t < 20; t += 8, ++ai) {
        int tr = t / 5, tc = t % 5;
#pragma unroll
        for (int r = 0; r < 8; ++r)
            og[(tc * 16 + ln) * 64 + tr * 16 + half * 8 + r] = acc[ai][r];
    }
}

// ---- wmat = pinv @ kern_qmk_v, f32 WMMA, output transposed bf16 -----------
__global__ __launch_bounds__(256) void pinv_apply_kernel(
    const float* __restrict__ pinvz, const float* __restrict__ kqv_t,
    bf16* __restrict__ wmat_t)
{
    int bh = blockIdx.x;
    __shared__ float P[64][68];    // pinv row-major (A-role)
    __shared__ float Qt[80][68];   // kqv transposed [c][m] (Bt-role)
    int tid = threadIdx.x, lane = tid & 31, wave = tid >> 5;
    int half = lane >> 4, ln = lane & 15;
    const float* pg = pinvz + (size_t)bh * 4096;
    const float* qg = kqv_t + (size_t)bh * D1P * 64;
    for (int e = tid; e < 1024; e += 256) {
        int r = e >> 4, q = e & 15;
        async_cp16(pg + r * 64 + q * 4, &P[r][q * 4]);
    }
    for (int e = tid; e < 1280; e += 256) {
        int r = e >> 4, q = e & 15;
        async_cp16(qg + r * 64 + q * 4, &Qt[r][q * 4]);
    }
    async_wait();
    __syncthreads();
    bf16* og = wmat_t + (size_t)bh * D1P * 64;
    for (int t = wave; t < 20; t += 8) {
        int tr = t / 5, tc = t % 5;
        v8f c = {};
#pragma unroll
        for (int k0 = 0; k0 < 64; k0 += 4) {
            v2f a = *reinterpret_cast<const v2f*>(&P[tr * 16 + ln][k0 + half * 2]);
            v2f b = *reinterpret_cast<const v2f*>(&Qt[tc * 16 + ln][k0 + half * 2]);
            c = __builtin_amdgcn_wmma_f32_16x16x4_f32(false, a, false, b, (short)0, c, false, false);
        }
#pragma unroll
        for (int r = 0; r < 8; ++r)
            og[(tc * 16 + ln) * 64 + tr * 16 + half * 8 + r] = (bf16)c[r];
    }
}

// ---- prod = activ(q @ k_m^T) @ wmat ; normalize ; head-interleave ---------
__global__ __launch_bounds__(256) void prod_kernel(
    const bf16* __restrict__ qb, const bf16* __restrict__ kmb,
    const bf16* __restrict__ wmat_t, bf16* __restrict__ ohb)
{
    int ch = blockIdx.x;       // N chunk of 64
    int bh = blockIdx.y;
    __shared__ bf16 qs[64][72];
    __shared__ bf16 kmB[64][72];   // k_m row-major == Bt of k_m^T
    __shared__ bf16 Sb[64][72];
    __shared__ bf16 Wt_l[80][72];  // wmat^T [c][m]: Bt of wmat
    __shared__ float Pf[64][84];
    int tid = threadIdx.x, lane = tid & 31, wave = tid >> 5;
    int half = lane >> 4, ln = lane & 15;

    const bf16* qg = qb + ((size_t)bh * NN + ch * 64) * DD;
    const bf16* kg = kmb + (size_t)bh * 4096;
    const bf16* wg = wmat_t + (size_t)bh * D1P * 64;
    for (int e = tid; e < 512; e += 256) {
        int r = e >> 3, q = e & 7;
        async_cp16(qg + r * 64 + q * 8, &qs[r][q * 8]);
        async_cp16(kg + r * 64 + q * 8, &kmB[r][q * 8]);
    }
    for (int e = tid; e < 640; e += 256) {
        int r = e >> 3, q = e & 7;
        async_cp16(wg + r * 64 + q * 8, &Wt_l[r][q * 8]);
    }
    async_wait();
    __syncthreads();

#pragma unroll
    for (int ti = 0; ti < 2; ++ti) {
        int t = wave * 2 + ti, tr = t >> 2, tc = t & 3;
        v8f c = {};
#pragma unroll
        for (int k0 = 0; k0 < 64; k0 += 32) {
            v16bf a = ldsA16(&qs[0][0], 72, tr * 16, k0, lane);
            v16bf b = ldsB16(&kmB[0][0], 72, tc * 16, k0, lane);
            c = __builtin_amdgcn_wmma_f32_16x16x32_bf16(false, a, false, b, (short)0, c, false, false);
        }
#pragma unroll
        for (int r = 0; r < 8; ++r)
            Sb[tr * 16 + half * 8 + r][tc * 16 + ln] = (bf16)activf(c[r]);
    }
    __syncthreads();

    for (int t = wave; t < 20; t += 8) {
        int tr = t / 5, tc = t % 5;
        v8f c = {};
#pragma unroll
        for (int k0 = 0; k0 < 64; k0 += 32) {
            v16bf a = ldsA16(&Sb[0][0], 72, tr * 16, k0, lane);
            v16bf b = ldsB16(&Wt_l[0][0], 72, tc * 16, k0, lane);
            c = __builtin_amdgcn_wmma_f32_16x16x32_bf16(false, a, false, b, (short)0, c, false, false);
        }
#pragma unroll
        for (int r = 0; r < 8; ++r)
            Pf[tr * 16 + half * 8 + r][tc * 16 + ln] = c[r];
    }
    __syncthreads();

    int b = bh >> 3, h = bh & 7;
    for (int e = tid; e < 4096; e += 256) {
        int r = e >> 6, d = e & 63;
        float val = Pf[r][d] / (Pf[r][64] + 1e-12f);
        ohb[((size_t)b * NN + ch * 64 + r) * CC + h * 64 + d] = (bf16)val;
    }
}

// ---------------------------------------------------------------------------

extern "C" void kernel_launch(void* const* d_in, const int* in_sizes, int n_in,
                              void* d_out, int out_size, void* d_ws, size_t ws_size,
                              hipStream_t stream) {
    const float* x   = (const float*)d_in[0];
    const float* Wq  = (const float*)d_in[1];
    const float* bq  = (const float*)d_in[2];
    const float* Wkv = (const float*)d_in[3];
    const float* bkv = (const float*)d_in[4];
    const float* Wp  = (const float*)d_in[5];
    const float* bp  = (const float*)d_in[6];
    float* out = (float*)d_out;

    char* ws = (char*)d_ws;
    size_t off = 0;
    auto alloc = [&](size_t bytes) { char* p = ws + off; off += (bytes + 255) & ~(size_t)255; return p; };

    bf16* xb     = (bf16*)alloc((size_t)BN * CC * 2);
    bf16* wqt    = (bf16*)alloc((size_t)CC * CC * 2);           // [n][k]
    bf16* wkvt   = (bf16*)alloc((size_t)2 * CC * CC * 2);       // [n][k]
    bf16* wpt    = (bf16*)alloc((size_t)CC * CC * 2);           // [n][k]
    bf16* qb     = (bf16*)alloc((size_t)BHN * NN * DD * 2);
    bf16* kb     = (bf16*)alloc((size_t)BHN * NN * DD * 2);
    bf16* v1t    = (bf16*)alloc((size_t)BHN * D1P * NN * 2);    // [bh][c][n]
    bf16* qmb    = (bf16*)alloc((size_t)BHN * MM * DD * 2);
    bf16* kmb    = (bf16*)alloc((size_t)BHN * MM * DD * 2);
    float* temp  = (float*)alloc((size_t)BHN * MM * MM * 4);
    float* pinvz = (float*)alloc((size_t)BHN * MM * MM * 4);
    float* kqv_t = (float*)alloc((size_t)BHN * D1P * 64 * 4);   // [bh][c][m]
    bf16* wmat_t = (bf16*)alloc((size_t)BHN * D1P * 64 * 2);    // [bh][c][m]
    bf16* ohb    = (bf16*)alloc((size_t)BN * CC * 2);
    unsigned* red = (unsigned*)alloc(256);

    zero_red_kernel<<<1, 1, 0, stream>>>(red);

    long nx = (long)BN * CC;
    f2bf_kernel<<<(unsigned)((nx + 255) / 256), 256, 0, stream>>>(x, xb, nx);
    f2bf_t_kernel<<<(CC * CC + 255) / 256, 256, 0, stream>>>(Wq, wqt, CC, CC);
    f2bf_t_kernel<<<(CC * 2 * CC + 255) / 256, 256, 0, stream>>>(Wkv, wkvt, CC, 2 * CC);
    f2bf_t_kernel<<<(CC * CC + 255) / 256, 256, 0, stream>>>(Wp, wpt, CC, CC);

    init_v1pad_kernel<<<(unsigned)(((long)BHN * 16 * NN + 255) / 256), 256, 0, stream>>>(v1t);

    gemm_proj<0><<<dim3(CC / 64, BN / 64), 256, 0, stream>>>(
        xb, wqt, bq, CC, qb, nullptr, nullptr, nullptr);
    gemm_proj<1><<<dim3(2 * CC / 64, BN / 64), 256, 0, stream>>>(
        xb, wkvt, bkv, CC, nullptr, kb, v1t, nullptr);

    long nlm = (long)BHN * MM * DD;
    pool_lm_kernel<<<(unsigned)((nlm + 255) / 256), 256, 0, stream>>>(qb, qmb);
    pool_lm_kernel<<<(unsigned)((nlm + 255) / 256), 256, 0, stream>>>(kb, kmb);

    temp_kernel<<<BHN, 256, 0, stream>>>(qmb, kmb, temp, red);

    pinv_kernel<<<BHN, 256, 6 * 64 * 68 * sizeof(float), stream>>>(
        temp, (const float*)red, pinvz);

    kqmv_kernel<<<BHN, 256, 0, stream>>>(qmb, kb, v1t, kqv_t);

    pinv_apply_kernel<<<BHN, 256, 0, stream>>>(pinvz, kqv_t, wmat_t);

    prod_kernel<<<dim3(NN / 64, BHN), 256, 0, stream>>>(qb, kmb, wmat_t, ohb);

    gemm_proj<2><<<dim3(CC / 64, BN / 64), 256, 0, stream>>>(
        ohb, wpt, bp, CC, nullptr, nullptr, nullptr, out);

    (void)in_sizes; (void)n_in; (void)out_size; (void)ws_size;
}
